// TGNMemory_59201829208560
// MI455X (gfx1250) — matrix operand order repaired and checked by reading, not compiled, not run on hardware
//
#include <hip/hip_runtime.h>
#include <hip/hip_bf16.h>
#include <stdint.h>

// ---------------- problem constants ----------------
#define BATCH     16384
#define MEM_DIM   1024
#define MSG_DIM   768
#define TIME_DIM  128
#define GRU_IN    2944            // 2*MEM + MSG + TIME
#define KBIG      3968            // GRU_IN + MEM_DIM (fused r/z K dimension)
#define NKSTEP    (KBIG / 32)     // 124 k-steps of 32
#define KSPLIT    (GRU_IN / 32)   // 92: k-step where gate-n switches ih -> hh

typedef __attribute__((ext_vector_type(16))) __bf16 v16bf;
typedef __attribute__((ext_vector_type(8)))  __bf16 v8bf;
typedef __attribute__((ext_vector_type(8)))  float  v8f;
typedef __attribute__((ext_vector_type(4)))  int    v4i;

#define AS1 __attribute__((address_space(1)))
#define AS3 __attribute__((address_space(3)))

#if __has_builtin(__builtin_amdgcn_global_load_async_to_lds_b128)
#define HAVE_ASYNC_LDS 1
#else
#define HAVE_ASYNC_LDS 0
#endif

// ---------------------------------------------------------------------------
// Kernel 1: gather + aggregate + time-encode one batch row into bf16 big-A.
// Row layout (width KBIG):
//   [0,1024)      src*d0 + dst*d1
//   [1024,2048)   src*d1 + dst*d0
//   [2048,2816)   raw_msg
//   [2816,2944)   cos(t * time_w + time_b)
//   [2944,3968)   src  (operand of the W_hh GEMM, fused into same K loop)
// ---------------------------------------------------------------------------
__global__ void __launch_bounds__(256)
prep_rows(const int* __restrict__ n_id, const int* __restrict__ memory_ints,
          const float* __restrict__ memory, const float* __restrict__ memory_msg,
          const float* __restrict__ direction, const float* __restrict__ time_w,
          const float* __restrict__ time_b,
          __bf16* __restrict__ A_big, float* __restrict__ out_last)
{
    const int m   = blockIdx.x;
    const int nid = n_id[m];
    const int last = memory_ints[3 * nid + 0];
    const int relt = memory_ints[3 * nid + 1];
    const int dst  = memory_ints[3 * nid + 2];
    const float d0 = direction[2 * nid + 0];
    const float d1 = direction[2 * nid + 1];

    const float* __restrict__ src  = memory     + (size_t)nid * MEM_DIM;
    const float* __restrict__ dstm = memory     + (size_t)dst * MEM_DIM;
    const float* __restrict__ msg  = memory_msg + (size_t)nid * MSG_DIM;
    __bf16* __restrict__ row = A_big + (size_t)m * KBIG;

    for (int j = threadIdx.x; j < MEM_DIM; j += 256) {
        float s = src[j], d = dstm[j];
        row[j]                 = (__bf16)(s * d0 + d * d1);
        row[MEM_DIM + j]       = (__bf16)(s * d1 + d * d0);
        row[GRU_IN + j]        = (__bf16)s;            // src tail for hh GEMM
    }
    for (int j = threadIdx.x; j < MSG_DIM; j += 256)
        row[2 * MEM_DIM + j] = (__bf16)msg[j];

    const float t = (float)relt;
    for (int j = threadIdx.x; j < TIME_DIM; j += 256)
        row[2 * MEM_DIM + MSG_DIM + j] = (__bf16)__cosf(t * time_w[j] + time_b[j]);

    if (threadIdx.x == 0)
        out_last[m] = (float)last;   // exactly representable (< 2^24)
}

// ---------------------------------------------------------------------------
// Kernel 2: weight conversion to bf16, K-major per output column.
//   Wr/Wz : [1024][KBIG]   k<2944 -> w_ih row, else w_hh row (fused K)
//   Wn_ih : [1024][2944],  Wn_hh : [1024][1024]
// ---------------------------------------------------------------------------
__global__ void __launch_bounds__(256)
prep_w(const float* __restrict__ w_ih, const float* __restrict__ w_hh,
       __bf16* __restrict__ Wr, __bf16* __restrict__ Wz,
       __bf16* __restrict__ Wn_ih, __bf16* __restrict__ Wn_hh)
{
    const int idx = blockIdx.x * 256 + threadIdx.x;

    if (idx < MEM_DIM * KBIG) {
        const int col = idx / KBIG;
        const int k   = idx % KBIG;
        float vr, vz;
        if (k < GRU_IN) {
            vr = w_ih[(size_t)col * GRU_IN + k];
            vz = w_ih[(size_t)(MEM_DIM + col) * GRU_IN + k];
        } else {
            vr = w_hh[(size_t)col * MEM_DIM + (k - GRU_IN)];
            vz = w_hh[(size_t)(MEM_DIM + col) * MEM_DIM + (k - GRU_IN)];
        }
        Wr[idx] = (__bf16)vr;
        Wz[idx] = (__bf16)vz;
    }
    if (idx < MEM_DIM * GRU_IN) {
        const int col = idx / GRU_IN;
        const int k   = idx % GRU_IN;
        Wn_ih[idx] = (__bf16)w_ih[(size_t)(2 * MEM_DIM + col) * GRU_IN + k];
    }
    if (idx < MEM_DIM * MEM_DIM) {
        const int col = idx >> 10;
        const int k   = idx & (MEM_DIM - 1);
        Wn_hh[idx] = (__bf16)w_hh[(size_t)(2 * MEM_DIM + col) * MEM_DIM + k];
    }
}

static __device__ inline float sigmoidf_fast(float x)
{
    return 1.0f / (1.0f + __expf(-x));
}

// 16-byte global -> LDS copy: async DMA on CDNA5.
static __device__ inline void copy16_to_lds(const __bf16* __restrict__ g,
                                            __bf16* __restrict__ l)
{
#if HAVE_ASYNC_LDS
    __builtin_amdgcn_global_load_async_to_lds_b128((AS1 v4i*)g, (AS3 v4i*)l, 0, 0);
#else
    *(v8bf*)l = *(const v8bf*)g;
#endif
}

static __device__ inline void wait_async_copies()
{
#if HAVE_ASYNC_LDS
#if __has_builtin(__builtin_amdgcn_s_wait_asynccnt)
    __builtin_amdgcn_s_wait_asynccnt(0);
#else
    asm volatile("s_wait_asynccnt 0x0" ::: "memory");
#endif
#endif
}

// two 8-half runs p[0..8) and p[16..24) from LDS, as the WMMA half-wave expects (8 + 8)
static __device__ inline v16bf lds_frag(const __bf16* p)
{
    v8bf lo = *(const v8bf*)(p);
    v8bf hi = *(const v8bf*)(p + 16);
    v16bf a;
#pragma unroll
    for (int i = 0; i < 8; ++i) { a[i] = lo[i]; a[8 + i] = hi[i]; }
    return a;
}

#define WMMA_BF16(a, b, c) \
    __builtin_amdgcn_wmma_f32_16x16x32_bf16(false, (a), false, (b), (short)0, (c), false, false)

// ---------------------------------------------------------------------------
// Kernel 3: fused GRU GEMM, all-async data movement.
//  Block = 256 thr = 8 waves arranged 2(M) x 4(N); block tile 64(M) x 64(N).
//  Each wave: 2 M-subtiles x 16 cols -> 8 f32 accumulators.
//  Per k-step the block stages A (64x32 = 4 KB) and B (3 gates x 64x32 =
//  12 KB) into LDS via GLOBAL_LOAD_ASYNC_TO_LDS_B128: exactly 4 branch-free
//  copies per thread (1 A + 1 per gate). Double buffered; the single
//  Wn_ih -> Wn_hh discontinuity is one uniform scalar pointer swap.
// ---------------------------------------------------------------------------
__global__ void __launch_bounds__(256, 2)
gru_fused(const __bf16* __restrict__ A_big,
          const __bf16* __restrict__ Wr, const __bf16* __restrict__ Wz,
          const __bf16* __restrict__ Wn_ih, const __bf16* __restrict__ Wn_hh,
          const float* __restrict__ b_ih, const float* __restrict__ b_hh,
          const int* __restrict__ n_id, const float* __restrict__ memory,
          float* __restrict__ out)
{
    __shared__ alignas(128) __bf16 smA[2][64 * 32];        // 2 x 4 KB
    __shared__ alignas(128) __bf16 smB[2][3 * 64 * 32];    // 2 x 12 KB

    const int tid  = threadIdx.x;
    const int lane = tid & 31;
    const int wv   = tid >> 5;
    const int mr   = wv >> 2;                 // M wave-row   0..1
    const int cg   = wv & 3;                  // N col-group  0..3
    const int rowBase = blockIdx.x * 64;
    const int colBase = blockIdx.y * 64;

    // ---- per-thread staging sources: advance +32 halfs per k-step ----
    const int scol = tid >> 2;                // 0..63 : B col / A row
    const int kseg = tid & 3;                 // 0..3  : 8-half (16 B) segment
    const __bf16* pa = A_big + (size_t)(rowBase + scol) * KBIG    + kseg * 8;
    const __bf16* pr = Wr    + (size_t)(colBase + scol) * KBIG    + kseg * 8;
    const __bf16* pz = Wz    + (size_t)(colBase + scol) * KBIG    + kseg * 8;
    const __bf16* pn = Wn_ih + (size_t)(colBase + scol) * GRU_IN  + kseg * 8;
    const __bf16* pn_hh = Wn_hh + (size_t)(colBase + scol) * MEM_DIM + kseg * 8;

    const int dstOff = scol * 32 + kseg * 8;  // halfs; per-gate region = 2048

    int s = 0;                                // next k-step index to stage
    auto stage = [&](int buf) {
        if (s == KSPLIT) pn = pn_hh;          // uniform scalar phase switch
        copy16_to_lds(pa, &smA[buf][dstOff]);
        copy16_to_lds(pr, &smB[buf][0 * 2048 + dstOff]);
        copy16_to_lds(pz, &smB[buf][1 * 2048 + dstOff]);
        copy16_to_lds(pn, &smB[buf][2 * 2048 + dstOff]);
        pa += 32; pr += 32; pz += 32; pn += 32;
        ++s;
    };

    v8f ar0 = {}, az0 = {}, ai0 = {}, ah0 = {};
    v8f ar1 = {}, az1 = {}, ai1 = {}, ah1 = {};

    stage(0);   // prologue

    // fragment offsets inside the LDS tiles (halfs)
    const int aOff0 = (mr * 32 + (lane & 15)) * 32 + ((lane >> 4) << 3);
    const int aOff1 = aOff0 + 16 * 32;
    const int bOff  = (cg * 16 + (lane & 15)) * 32 + ((lane >> 4) << 3);

    // ---- phase 1: k-steps 0..91 (gate-n accumulates i_n) ----
    for (int ks = 0; ks < KSPLIT; ++ks) {
        const int cur = ks & 1;
        wait_async_copies();
        __syncthreads();
        stage(cur ^ 1);                        // always valid here

        const __bf16* bA = &smA[cur][0];
        const __bf16* bB = &smB[cur][0];
        v16bf a0 = lds_frag(bA + aOff0);
        v16bf a1 = lds_frag(bA + aOff1);
        v16bf br = lds_frag(bB + 0 * 2048 + bOff);
        v16bf bz = lds_frag(bB + 1 * 2048 + bOff);
        v16bf bn = lds_frag(bB + 2 * 2048 + bOff);

        ar0 = WMMA_BF16(a0, br, ar0);
        ar1 = WMMA_BF16(a1, br, ar1);
        az0 = WMMA_BF16(a0, bz, az0);
        az1 = WMMA_BF16(a1, bz, az1);
        ai0 = WMMA_BF16(a0, bn, ai0);
        ai1 = WMMA_BF16(a1, bn, ai1);
    }

    // ---- phase 2: k-steps 92..123 (gate-n accumulates h_n) ----
    for (int ks = KSPLIT; ks < NKSTEP; ++ks) {
        const int cur = ks & 1;
        wait_async_copies();
        __syncthreads();
        if (ks + 1 < NKSTEP)                   // uniform tail guard
            stage(cur ^ 1);

        const __bf16* bA = &smA[cur][0];
        const __bf16* bB = &smB[cur][0];
        v16bf a0 = lds_frag(bA + aOff0);
        v16bf a1 = lds_frag(bA + aOff1);
        v16bf br = lds_frag(bB + 0 * 2048 + bOff);
        v16bf bz = lds_frag(bB + 1 * 2048 + bOff);
        v16bf bn = lds_frag(bB + 2 * 2048 + bOff);

        ar0 = WMMA_BF16(a0, br, ar0);
        ar1 = WMMA_BF16(a1, br, ar1);
        az0 = WMMA_BF16(a0, bz, az0);
        az1 = WMMA_BF16(a1, bz, az1);
        ah0 = WMMA_BF16(a0, bn, ah0);
        ah1 = WMMA_BF16(a1, bn, ah1);
    }

    // Epilogue. C/D layout: VGPR i, lane l -> N = l%16, M = i + 8*(l/16).
    const int col = colBase + cg * 16 + (lane & 15);
    const float bias_r  = b_ih[col]           + b_hh[col];
    const float bias_z  = b_ih[MEM_DIM + col] + b_hh[MEM_DIM + col];
    const float bias_in = b_ih[2 * MEM_DIM + col];
    const float bias_hn = b_hh[2 * MEM_DIM + col];

#pragma unroll
    for (int t = 0; t < 2; ++t) {
        const v8f& accr = t ? ar1 : ar0;
        const v8f& accz = t ? az1 : az0;
        const v8f& acci = t ? ai1 : ai0;
        const v8f& acch = t ? ah1 : ah0;
#pragma unroll
        for (int i = 0; i < 8; ++i) {
            const int row = rowBase + mr * 32 + t * 16 + i + ((lane >> 4) << 3);
            const float r = sigmoidf_fast(accr[i] + bias_r);
            const float z = sigmoidf_fast(accz[i] + bias_z);
            const float n = tanhf(acci[i] + bias_in + r * (acch[i] + bias_hn));
            const float sm = memory[(size_t)n_id[row] * MEM_DIM + col];
            out[(size_t)row * MEM_DIM + col] = (1.0f - z) * n + z * sm;
        }
    }
}

// ---------------------------------------------------------------------------
extern "C" void kernel_launch(void* const* d_in, const int* in_sizes, int n_in,
                              void* d_out, int out_size, void* d_ws, size_t ws_size,
                              hipStream_t stream)
{
    const int*   n_id        = (const int*)  d_in[0];
    const int*   memory_ints = (const int*)  d_in[1];
    const float* memory      = (const float*)d_in[2];
    const float* memory_msg  = (const float*)d_in[3];
    const float* direction   = (const float*)d_in[4];
    const float* time_w      = (const float*)d_in[5];
    const float* time_b      = (const float*)d_in[6];
    const float* gru_w_ih    = (const float*)d_in[7];
    const float* gru_w_hh    = (const float*)d_in[8];
    const float* gru_b_ih    = (const float*)d_in[9];
    const float* gru_b_hh    = (const float*)d_in[10];

    float* out_mem  = (float*)d_out;                       // 16384 x 1024
    float* out_last = out_mem + (size_t)BATCH * MEM_DIM;   // 16384

    // workspace partition (bf16):
    char* ws = (char*)d_ws;
    __bf16* A_big = (__bf16*)ws;                            ws += (size_t)BATCH * KBIG * 2;
    __bf16* Wr    = (__bf16*)ws;                            ws += (size_t)MEM_DIM * KBIG * 2;
    __bf16* Wz    = (__bf16*)ws;                            ws += (size_t)MEM_DIM * KBIG * 2;
    __bf16* Wn_ih = (__bf16*)ws;                            ws += (size_t)MEM_DIM * GRU_IN * 2;
    __bf16* Wn_hh = (__bf16*)ws;

    prep_rows<<<BATCH, 256, 0, stream>>>(n_id, memory_ints, memory, memory_msg,
                                         direction, time_w, time_b, A_big, out_last);

    const int wtotal = MEM_DIM * KBIG;
    prep_w<<<(wtotal + 255) / 256, 256, 0, stream>>>(gru_w_ih, gru_w_hh,
                                                     Wr, Wz, Wn_ih, Wn_hh);

    dim3 grid(BATCH / 64, MEM_DIM / 64);
    gru_fused<<<grid, 256, 0, stream>>>(A_big, Wr, Wz, Wn_ih, Wn_hh,
                                        gru_b_ih, gru_b_hh, n_id, memory, out_mem);
}
